// AIAConv_46943992545630
// MI455X (gfx1250) — matrix operand order repaired and checked by reading, not compiled, not run on hardware
//
#include <hip/hip_runtime.h>
#include <stdint.h>

// ---------------- WMMA types / helpers (gfx1250, wave32) ----------------
typedef __attribute__((ext_vector_type(4)))  unsigned int v4u;
typedef __attribute__((ext_vector_type(16))) __bf16       v16bf;
typedef __attribute__((ext_vector_type(8)))  float        v8f;

union Frag { v16bf v; v4u u[2]; };

__device__ __forceinline__ unsigned short f2bf(float x) {
    union { float f; unsigned int u; } c; c.f = x;
    unsigned int u = c.u;
    u += 0x7fffu + ((u >> 16) & 1u);   // round-to-nearest-even
    return (unsigned short)(u >> 16);
}
__device__ __forceinline__ float bf2f(unsigned short h) {
    union { float f; unsigned int u; } c; c.u = ((unsigned int)h) << 16;
    return c.f;
}

__device__ __forceinline__ v8f wmma_bf16(const Frag& a, const Frag& b, v8f c) {
    return __builtin_amdgcn_wmma_f32_16x16x32_bf16(
        false, a.v, false, b.v, (short)0, c, false, false);
}

// ---------------- constants ----------------
#define Bq   16
#define Sq   512
#define DIN  512
#define DOUT 512
#define Hh   8
#define Mm   4
#define HFf  64
#define NHB  128          // H*B
#define ATT_LDA 1024      // bf16 stride of in-place attn rows inside f32 score buffer

// Cooperative async stage of a 64KB bf16 panel (64 x 512) into LDS.
// Uses GLOBAL_LOAD_ASYNC_TO_LDS_B128 (ASYNCcnt) then s_wait_asynccnt + barrier.
__device__ __forceinline__ void stage_panel_async(const unsigned short* src,
                                                  unsigned short* dst_lds, int tid) {
    uint64_t gbase = (uint64_t)(uintptr_t)src;
    uint32_t lbase = (uint32_t)(uintptr_t)dst_lds;   // low 32 bits of LDS aperture addr = LDS offset
#pragma unroll
    for (int i = 0; i < 16; ++i) {                   // 4096 16B chunks / 256 threads
        uint32_t chunk = (uint32_t)(i * 256 + tid);
        uint64_t ga = gbase + (uint64_t)chunk * 16u;
        uint32_t la = lbase + chunk * 16u;
        asm volatile("global_load_async_to_lds_b128 %0, %1, off"
                     :: "v"(la), "v"(ga) : "memory");
    }
    asm volatile("s_wait_asynccnt 0x0" ::: "memory");
    __syncthreads();
}

// ---------------- prep kernels ----------------
__global__ void k_prep_w(const float* Wq, const float* Wk, const float* Wv,
                         const float* Wt, unsigned short* WT) {
    int id = blockIdx.x * 256 + threadIdx.x;            // 4*512*512
    int w = id >> 18, rem = id & 262143;
    int k = rem >> 9, n = rem & 511;
    const float* W = (w == 0) ? Wq : (w == 1) ? Wk : (w == 2) ? Wv : Wt;
    WT[(size_t)w * 262144 + (size_t)n * 512 + k] = f2bf(W[(size_t)k * 512 + n]);
}

__global__ void k_prep_wi(const float* Wi, unsigned short* WiT) {
    int id = blockIdx.x * 256 + threadIdx.x;            // 256*64
    int n = id >> 6, k = id & 63;
    WiT[(size_t)n * 64 + k] = f2bf(Wi[(size_t)k * 256 + n]);
}

__global__ void k_convert(const float* q, const float* k,
                          unsigned short* qb, unsigned short* kb) {
    int id = blockIdx.x * 256 + threadIdx.x;            // 8192*512
    qb[id] = f2bf(q[id]);
    kb[id] = f2bf(k[id]);
}

// ---------------- projections (1x4 register-blocked over one head) ----------------
// Q,K row-major [hb,S,64]; V,T transposed [hb,64,S]
__global__ void k_proj(const unsigned short* qb, const unsigned short* kb,
                       const unsigned short* WT,
                       unsigned short* Qh, unsigned short* Kh,
                       unsigned short* Vt, unsigned short* Tt) {
    int wid  = blockIdx.x * 8 + (threadIdx.x >> 5);     // 16384 waves
    int lane = threadIdx.x & 31;
    int which = wid >> 12;                              // 4096 waves per matrix
    int t = wid & 4095;
    int m0 = (t >> 3) << 4;                             // 512 m-tiles
    int ng = t & 7;                                     // head index; n = ng*64 + j*16 + ln
    const unsigned short* A  = (which == 0) ? qb : kb;
    const unsigned short* Bt = WT + (size_t)which * 262144;

    int ln = lane & 15, hi = lane >> 4;
    const unsigned short* arow = A  + (size_t)(m0 + ln) * DIN + (hi << 3);
    const unsigned short* brow = Bt + (size_t)(ng * 64 + ln) * DIN + (hi << 3);

    v8f vzero = {};
    v8f acc[4] = {vzero, vzero, vzero, vzero};
#pragma unroll
    for (int k0 = 0; k0 < DIN; k0 += 32) {
        Frag a;
        a.u[0] = *(const v4u*)(arow + k0);
        a.u[1] = *(const v4u*)(arow + k0 + 16);
#pragma unroll
        for (int j = 0; j < 4; ++j) {
            Frag b;
            const unsigned short* bp = brow + (size_t)j * 16 * DIN + k0;
            b.u[0] = *(const v4u*)(bp);
            b.u[1] = *(const v4u*)(bp + 16);
            acc[j] = wmma_bf16(a, b, acc[j]);
        }
    }
#pragma unroll
    for (int j = 0; j < 4; ++j) {
        int f = j * 16 + ln;                            // feature within head
#pragma unroll
        for (int v = 0; v < 8; ++v) {
            int m = m0 + v + (hi << 3);
            int b = m >> 9, s = m & 511;
            int hb = ng * Bq + b;
            unsigned short val = f2bf(acc[j][v]);
            if (which == 0)      Qh[((size_t)hb * Sq + s) * HFf + f] = val;
            else if (which == 1) Kh[((size_t)hb * Sq + s) * HFf + f] = val;
            else if (which == 2) Vt[((size_t)hb * HFf + f) * Sq + s] = val;
            else                 Tt[((size_t)hb * HFf + f) * Sq + s] = val;
        }
    }
}

// ---------------- scores = Q K^T / sqrt(512), causal mask (A-frags reused x4) ----------------
__global__ void k_scores(const unsigned short* Qh, const unsigned short* Kh, float* scores) {
    int wid  = blockIdx.x * 8 + (threadIdx.x >> 5);     // 32768 waves
    int lane = threadIdx.x & 31;
    int hb  = wid >> 8;                                 // 256 waves per hb
    int t   = wid & 255;
    int m0  = (t >> 3) << 4;                            // q tile
    int ktg = t & 7;                                    // key-tile group of 4
    int ln = lane & 15, hi = lane >> 4;

    const unsigned short* arow = Qh + (size_t)hb * Sq * HFf + (size_t)(m0 + ln) * HFf + (hi << 3);
    Frag a0, a1;
    a0.u[0] = *(const v4u*)(arow);       a0.u[1] = *(const v4u*)(arow + 16);
    a1.u[0] = *(const v4u*)(arow + 32);  a1.u[1] = *(const v4u*)(arow + 48);

    const unsigned short* Bt = Kh + (size_t)hb * Sq * HFf;
    float* srow = scores + (size_t)hb * Sq * Sq;
#pragma unroll
    for (int j = 0; j < 4; ++j) {
        int n0 = ktg * 64 + j * 16;
        const unsigned short* brow = Bt + (size_t)(n0 + ln) * HFf + (hi << 3);
        Frag b0, b1;
        b0.u[0] = *(const v4u*)(brow);       b0.u[1] = *(const v4u*)(brow + 16);
        b1.u[0] = *(const v4u*)(brow + 32);  b1.u[1] = *(const v4u*)(brow + 48);
        v8f acc = {};
        acc = wmma_bf16(a0, b0, acc);
        acc = wmma_bf16(a1, b1, acc);
        int kcol = n0 + ln;
#pragma unroll
        for (int v = 0; v < 8; ++v) {
            int q = m0 + v + (hi << 3);
            float val = acc[v] * 0.04419417382415922f;  // 1/sqrt(512)
            if (kcol > q) val = -4294967295.0f;         // -2^32 + 1
            srow[(size_t)q * Sq + kcol] = val;
        }
    }
}

// ---------------- softmax per row; write bf16 attn in place (lda=1024 halves) ----------------
__global__ void k_softmax(float* scores) {
    int row = blockIdx.x;                               // hb*512 + q
    float* srow = scores + (size_t)row * Sq;
    int t = threadIdx.x;
    float x0 = srow[t], x1 = srow[t + 256];
    __shared__ float red[256];
    red[t] = fmaxf(x0, x1);
    __syncthreads();
    for (int s = 128; s > 0; s >>= 1) { if (t < s) red[t] = fmaxf(red[t], red[t + s]); __syncthreads(); }
    float mx = red[0];
    __syncthreads();
    float e0 = __expf(x0 - mx), e1 = __expf(x1 - mx);
    red[t] = e0 + e1;
    __syncthreads();
    for (int s = 128; s > 0; s >>= 1) { if (t < s) red[t] += red[t + s]; __syncthreads(); }
    float inv = 1.0f / red[0];
    __syncthreads();                                    // all f32 reads done before aliasing writes
    unsigned short* arow = (unsigned short*)srow;       // first 1KB of the 2KB row
    arow[t]       = f2bf(e0 * inv);
    arow[t + 256] = f2bf(e1 * inv);
}

// ---------------- E = attn @ T (B panel staged in LDS via async copy) ----------------
__global__ void k_E(const unsigned short* attn, const unsigned short* Tt, unsigned short* Eb) {
    __shared__ unsigned short panel[HFf * Sq];          // 64KB
    int hb  = blockIdx.x >> 1;
    int hlf = blockIdx.x & 1;
    int tid = threadIdx.x;
    stage_panel_async(Tt + (size_t)hb * HFf * Sq, panel, tid);

    int wave = tid >> 5, lane = tid & 31;
    int ln = lane & 15, hi = lane >> 4;
    const unsigned short* Abase = attn + (size_t)hb * Sq * ATT_LDA;
    const unsigned short* bbase = panel + (size_t)ln * Sq + (hi << 3);

#pragma unroll
    for (int qi = 0; qi < 2; ++qi) {
        int m0 = (hlf * 16 + wave * 2 + qi) << 4;
        const unsigned short* arow = Abase + (size_t)(m0 + ln) * ATT_LDA + (hi << 3);
        v8f vzero = {};
        v8f acc[4] = {vzero, vzero, vzero, vzero};
#pragma unroll
        for (int k0 = 0; k0 < Sq; k0 += 32) {
            Frag a;
            a.u[0] = *(const v4u*)(arow + k0);
            a.u[1] = *(const v4u*)(arow + k0 + 16);
#pragma unroll
            for (int j = 0; j < 4; ++j) {
                Frag b;                                  // ds_load_b128 x2 from LDS
                const unsigned short* bp = bbase + (size_t)j * 16 * Sq + k0;
                b.u[0] = *(const v4u*)(bp);
                b.u[1] = *(const v4u*)(bp + 16);
                acc[j] = wmma_bf16(a, b, acc[j]);
            }
        }
#pragma unroll
        for (int v = 0; v < 8; ++v) {
            int q = m0 + v + (hi << 3);
            unsigned short* erow = Eb + ((size_t)hb * Sq + q) * HFf + ln;
            erow[0]  = f2bf(acc[0][v]);
            erow[16] = f2bf(acc[1][v]);
            erow[32] = f2bf(acc[2][v]);
            erow[48] = f2bf(acc[3][v]);
        }
    }
}

// ---------------- intensities: mu = sigmoid([E,ts]@Wi + bi); lam = scaled softplus ----------------
__global__ void k_intensity(const unsigned short* Eb, const unsigned short* WiT,
                            const float* Wi, const float* bi,
                            const float* weight_i, const float* scale_i,
                            const float* tsp, float* lam_out) {
    int wid  = blockIdx.x * 8 + (threadIdx.x >> 5);     // 32768 m-tiles
    int lane = threadIdx.x & 31;
    int m0 = wid << 4;
    int ln = lane & 15, hi = lane >> 4;

    const unsigned short* arow = Eb + (size_t)(m0 + ln) * HFf + (hi << 3);
    Frag a0, a1;
    a0.u[0] = *(const v4u*)(arow);       a0.u[1] = *(const v4u*)(arow + 16);
    a1.u[0] = *(const v4u*)(arow + 32);  a1.u[1] = *(const v4u*)(arow + 48);

    float ts[8];
#pragma unroll
    for (int v = 0; v < 8; ++v) {
        int r = m0 + v + (hi << 3);                     // r = hb*512 + s
        ts[v] = tsp[(((r >> 9) & 15) << 9) + (r & 511)];
    }
    const unsigned short* bbase = WiT + (size_t)ln * 64 + (hi << 3);

    for (int mark = 0; mark < Mm; ++mark) {
        float part[8] = {0, 0, 0, 0, 0, 0, 0, 0};
#pragma unroll
        for (int nt = 0; nt < 4; ++nt) {
            int n0 = mark * 64 + nt * 16;
            const unsigned short* brow = bbase + (size_t)n0 * 64;
            Frag b0, b1;
            b0.u[0] = *(const v4u*)(brow);       b0.u[1] = *(const v4u*)(brow + 16);
            b1.u[0] = *(const v4u*)(brow + 32);  b1.u[1] = *(const v4u*)(brow + 48);
            v8f acc = {};
            acc = wmma_bf16(a0, b0, acc);
            acc = wmma_bf16(a1, b1, acc);
            int n = n0 + ln;                            // column in [0,256)
            float wl = Wi[64 * 256 + n];                // ts row of fc_i weight
            float bn = bi[n];
            float wi_n = weight_i[n];                   // [M,HF] flat == n
#pragma unroll
            for (int v = 0; v < 8; ++v) {
                float mu = 1.0f / (1.0f + __expf(-(acc[v] + ts[v] * wl + bn)));
                part[v] += mu * wi_n;
            }
        }
        for (int off = 1; off < 16; off <<= 1)
#pragma unroll
            for (int v = 0; v < 8; ++v)
                part[v] += __shfl_xor(part[v], off, 32);
        if (ln == 0) {
            float sc = __expf(scale_i[mark]);
#pragma unroll
            for (int v = 0; v < 8; ++v) {
                int r = m0 + v + (hi << 3);
                float x = part[v] / sc;
                lam_out[(size_t)r * Mm + mark] = sc * log1pf(__expf(x));
            }
        }
    }
}

// ---------------- attnm = attn * (lam . em) ----------------
__global__ void k_attnm(const unsigned short* attn, unsigned short* attnm,
                        const float* lam, const float* em) {
    int id = blockIdx.x * 256 + threadIdx.x;            // 65536 rows * 64 groups
    int kg = id & 63;
    int rq = id >> 6;                                   // hb*512 + q
    const float4 l4 = *(const float4*)(lam + (size_t)rq * 4);
    int hb = rq >> 9;
    const unsigned short* arow = attn  + (size_t)rq * ATT_LDA + kg * 8;
    unsigned short*       orow = attnm + (size_t)rq * ATT_LDA + kg * 8;
    const float* emb = em + ((size_t)hb * Sq) * 4;
    union { unsigned short s[8]; v4u u; } o;
#pragma unroll
    for (int j = 0; j < 8; ++j) {
        int kk = kg * 8 + j;
        const float4 e4 = *(const float4*)(emb + (size_t)kk * 4);
        float mi = l4.x * e4.x + l4.y * e4.y + l4.z * e4.z + l4.w * e4.w;
        o.s[j] = f2bf(bf2f(arow[j]) * mi);
    }
    *(v4u*)orow = o.u;
}

// ---------------- out = attnm @ V + queries (V panel staged in LDS) ----------------
__global__ void k_out(const unsigned short* attnm, const unsigned short* Vt,
                      const float* queries, float* out) {
    __shared__ unsigned short panel[HFf * Sq];          // 64KB
    int hb  = blockIdx.x >> 1;
    int hlf = blockIdx.x & 1;
    int tid = threadIdx.x;
    stage_panel_async(Vt + (size_t)hb * HFf * Sq, panel, tid);

    int wave = tid >> 5, lane = tid & 31;
    int ln = lane & 15, hi = lane >> 4;
    const unsigned short* Abase = attnm + (size_t)hb * Sq * ATT_LDA;
    const unsigned short* bbase = panel + (size_t)ln * Sq + (hi << 3);
    int h = hb >> 4, b = hb & 15;

#pragma unroll
    for (int qi = 0; qi < 2; ++qi) {
        int m0 = (hlf * 16 + wave * 2 + qi) << 4;
        const unsigned short* arow = Abase + (size_t)(m0 + ln) * ATT_LDA + (hi << 3);
        v8f vzero = {};
        v8f acc[4] = {vzero, vzero, vzero, vzero};
#pragma unroll
        for (int k0 = 0; k0 < Sq; k0 += 32) {
            Frag a;
            a.u[0] = *(const v4u*)(arow + k0);
            a.u[1] = *(const v4u*)(arow + k0 + 16);
#pragma unroll
            for (int j = 0; j < 4; ++j) {
                Frag bfr;
                const unsigned short* bp = bbase + (size_t)j * 16 * Sq + k0;
                bfr.u[0] = *(const v4u*)(bp);
                bfr.u[1] = *(const v4u*)(bp + 16);
                acc[j] = wmma_bf16(a, bfr, acc[j]);
            }
        }
#pragma unroll
        for (int v = 0; v < 8; ++v) {
            int q = m0 + v + (hi << 3);
            size_t base = ((size_t)(b * Sq + q)) * DOUT + h * HFf + ln;
            out[base]      = acc[0][v] + queries[base];
            out[base + 16] = acc[1][v] + queries[base + 16];
            out[base + 32] = acc[2][v] + queries[base + 32];
            out[base + 48] = acc[3][v] + queries[base + 48];
        }
    }
}

// ---------------- host launch ----------------
extern "C" void kernel_launch(void* const* d_in, const int* in_sizes, int n_in,
                              void* d_out, int out_size, void* d_ws, size_t ws_size,
                              hipStream_t stream) {
    (void)in_sizes; (void)n_in; (void)out_size; (void)ws_size;
    const float* queries  = (const float*)d_in[0];
    const float* keys     = (const float*)d_in[1];
    const float* tspans   = (const float*)d_in[2];
    // d_in[3] attention_masks: broadcast tril -> applied analytically (kcol > q)
    const float* em       = (const float*)d_in[4];
    const float* Wq       = (const float*)d_in[5];
    const float* Wk       = (const float*)d_in[6];
    const float* Wv       = (const float*)d_in[7];
    const float* Wt       = (const float*)d_in[8];
    const float* Wi       = (const float*)d_in[9];
    const float* bi       = (const float*)d_in[10];
    const float* weight_i = (const float*)d_in[11];
    const float* scale_i  = (const float*)d_in[12];

    char* ws = (char*)d_ws;
    unsigned short* qb  = (unsigned short*)(ws + 0);          //  8,388,608 B
    unsigned short* kb  = (unsigned short*)(ws + 8388608);    //  8,388,608 B
    unsigned short* WT  = (unsigned short*)(ws + 16777216);   //  2,097,152 B
    unsigned short* WiT = (unsigned short*)(ws + 18874368);   //     32,768 B
    unsigned short* Qh  = (unsigned short*)(ws + 18907136);   //  8,388,608 B
    unsigned short* Kh  = (unsigned short*)(ws + 27295744);   //  8,388,608 B
    unsigned short* Vt  = (unsigned short*)(ws + 35684352);   //  8,388,608 B
    unsigned short* Tt  = (unsigned short*)(ws + 44072960);   //  8,388,608 B
    unsigned short* Eb  = (unsigned short*)(ws + 52461568);   //  8,388,608 B
    float*          sc  = (float*)(ws + 60850176);            // 134,217,728 B (reused: attn/attnm bf16)
    // total: 195,067,904 bytes

    float* out = (float*)d_out;                               // [16,512,512]
    float* lam = out + (size_t)Bq * Sq * DOUT;                // all_mark_inty [128,512,4]

    unsigned short* attn  = (unsigned short*)sc;              // lda = 1024 halves
    unsigned short* attnm = attn + 512;                       // second half of each row

    k_prep_w   <<<4096,  256, 0, stream>>>(Wq, Wk, Wv, Wt, WT);
    k_prep_wi  <<<64,    256, 0, stream>>>(Wi, WiT);
    k_convert  <<<16384, 256, 0, stream>>>(queries, keys, qb, kb);
    k_proj     <<<2048,  256, 0, stream>>>(qb, kb, WT, Qh, Kh, Vt, Tt);
    k_scores   <<<4096,  256, 0, stream>>>(Qh, Kh, sc);
    k_softmax  <<<65536, 256, 0, stream>>>(sc);
    k_E        <<<256,   256, 0, stream>>>(attn, Tt, Eb);
    k_intensity<<<4096,  256, 0, stream>>>(Eb, WiT, Wi, bi, weight_i, scale_i, tspans, lam);
    k_attnm    <<<16384, 256, 0, stream>>>(attn, attnm, lam, em);
    k_out      <<<256,   256, 0, stream>>>(attnm, Vt, queries, out);
}
